// CausalSelfAttention_22067541967504
// MI455X (gfx1250) — compile-verified, matrix operand order checked by reading, not run on hardware
//
#include <hip/hip_runtime.h>
#include <hip/hip_bf16.h>

// ---------------------------------------------------------------------------
// Types for CDNA5 WMMA (wave32): v_wmma_f32_16x16x32_bf16
// ---------------------------------------------------------------------------
typedef __bf16 bf16_t;
typedef __attribute__((ext_vector_type(16))) bf16_t bf16x16;
typedef __attribute__((ext_vector_type(8)))  float  f32x8;

union FragBF {
  bf16x16 v;
  uint4   q[2];
  unsigned short u[16];
};

__device__ __forceinline__ unsigned short f2bf(float f) {
  unsigned int u = __float_as_uint(f);
  unsigned int r = u + 0x7FFFu + ((u >> 16) & 1u);   // RNE
  return (unsigned short)(r >> 16);
}

// ---------------------------------------------------------------------------
// fp32 -> bf16 convert (vectorized, n multiple of 4)
// ---------------------------------------------------------------------------
__global__ __launch_bounds__(256) void cvt_f32_bf16(
    const float* __restrict__ in, unsigned short* __restrict__ out, int n) {
  int i = (blockIdx.x * 256 + threadIdx.x) * 4;
  if (i >= n) return;
  float4 f = *(const float4*)(in + i);
  unsigned short r[4];
  r[0] = f2bf(f.x); r[1] = f2bf(f.y); r[2] = f2bf(f.z); r[3] = f2bf(f.w);
  *(uint2*)(out + i) = *(uint2*)r;
}

// ---------------------------------------------------------------------------
// Tiled bf16 WMMA GEMM: C[M,N] = A[M,K] @ W[K,N] + bias
//   Block tile 128x128, BK=64, 256 threads = 8 waves (4x2), wave tile 32x64.
//   Register double-buffered staging: next tile's global loads are issued
//   before the WMMA block so HBM/L2 latency hides behind 16 WMMAs.
//   MODE 0: scatter qkv -> q[B,H,T,D], k[B,H,T,D], vT[B,H,D,T]  (bf16)
//   MODE 1: fp32 row-major output
// ---------------------------------------------------------------------------
template<int N, int K, int MODE>
__global__ __launch_bounds__(256) void gemm_bf16(
    const unsigned short* __restrict__ A,
    const unsigned short* __restrict__ W,
    const float* __restrict__ bias,
    unsigned short* __restrict__ oq,
    unsigned short* __restrict__ ok,
    unsigned short* __restrict__ ovt,
    float* __restrict__ outf) {
  constexpr int BM = 128, BN = 128, BK = 64, LS = 72;  // 144B rows (16B aligned)
  __shared__ unsigned short As[BM * LS];
  __shared__ unsigned short Bs[BN * LS];   // stored transposed: Bs[n][k]

  const int tid  = threadIdx.x;
  const int lane = tid & 31;
  const int wave = tid >> 5;
  const int wr   = wave >> 1;      // 0..3 (row group of 32)
  const int wc   = wave & 1;       // 0..1 (col group of 64)
  const int ln15 = lane & 15;
  const int hi   = lane >> 4;      // 0/1
  const int kb   = hi * 8;         // per-lane K base inside fragment

  const int bn0 = blockIdx.x * BN;
  const int bm0 = blockIdx.y * BM;

  // staging coordinates
  const int ar = tid >> 3;          // A row (+ i*32)
  const int ac = (tid & 7) * 8;     // A col chunk
  const int wk = tid >> 4;          // W k row (+ i*16)
  const int wn = (tid & 15) * 8;    // W n chunk

  const unsigned short* Aptr = A + (size_t)(bm0 + ar) * K + ac;
  const unsigned short* Wptr = W + (size_t)wk * N + bn0 + wn;

  f32x8 zero = {};
  f32x8 acc[2][4];
#pragma unroll
  for (int r = 0; r < 2; ++r)
#pragma unroll
    for (int c = 0; c < 4; ++c) acc[r][c] = zero;

  uint4 ra[4], rb[4];
#pragma unroll
  for (int i = 0; i < 4; ++i) ra[i] = *(const uint4*)(Aptr + (size_t)i * 32 * K);
#pragma unroll
  for (int i = 0; i < 4; ++i) rb[i] = *(const uint4*)(Wptr + (size_t)i * 16 * N);

  for (int k0 = 0; k0 < K; k0 += BK) {
    // ---- commit staged registers to LDS
#pragma unroll
    for (int i = 0; i < 4; ++i)
      *(uint4*)(&As[(ar + i * 32) * LS + ac]) = ra[i];
#pragma unroll
    for (int i = 0; i < 4; ++i) {
      const unsigned short* ds = (const unsigned short*)&rb[i];
      int kr = wk + i * 16;
#pragma unroll
      for (int j = 0; j < 8; ++j) Bs[(wn + j) * LS + kr] = ds[j];
    }
    __syncthreads();

    // ---- prefetch next tile into registers (overlaps with WMMA below)
    if (k0 + BK < K) {
#pragma unroll
      for (int i = 0; i < 4; ++i)
        ra[i] = *(const uint4*)(Aptr + (k0 + BK) + (size_t)i * 32 * K);
#pragma unroll
      for (int i = 0; i < 4; ++i)
        rb[i] = *(const uint4*)(Wptr + (size_t)(k0 + BK) * N + (size_t)i * 16 * N);
    }

    // ---- 2 K-steps x 8 WMMA
#pragma unroll
    for (int ks = 0; ks < 2; ++ks) {
      FragBF a[2], b[4];
#pragma unroll
      for (int r = 0; r < 2; ++r) {
        const unsigned short* p = &As[(wr * 32 + r * 16 + ln15) * LS + ks * 32 + kb];
        a[r].q[0] = *(const uint4*)(p);
        a[r].q[1] = *(const uint4*)(p + 16);
      }
#pragma unroll
      for (int c = 0; c < 4; ++c) {
        const unsigned short* p = &Bs[(wc * 64 + c * 16 + ln15) * LS + ks * 32 + kb];
        b[c].q[0] = *(const uint4*)(p);
        b[c].q[1] = *(const uint4*)(p + 16);
      }
#pragma unroll
      for (int r = 0; r < 2; ++r)
#pragma unroll
        for (int c = 0; c < 4; ++c)
          acc[r][c] = __builtin_amdgcn_wmma_f32_16x16x32_bf16(
              false, a[r].v, false, b[c].v, (short)0, acc[r][c], false, false);
    }
    __syncthreads();
  }

  // ---- epilogue
#pragma unroll
  for (int r = 0; r < 2; ++r) {
#pragma unroll
    for (int c = 0; c < 4; ++c) {
#pragma unroll
      for (int e = 0; e < 8; ++e) {
        int m = bm0 + wr * 32 + r * 16 + e + hi * 8;
        int n = bn0 + wc * 64 + c * 16 + ln15;
        float v = acc[r][c][e] + bias[n];
        if (MODE == 0) {
          // qkv scatter: T=2048, H=16, D=64
          int which = n >> 10;
          int cn = n & 1023;
          int h = cn >> 6, d = cn & 63;
          int bb = m >> 11, t = m & 2047;
          if (which == 0)
            oq[(((size_t)(bb * 16 + h)) * 2048 + t) * 64 + d] = f2bf(v);
          else if (which == 1)
            ok[(((size_t)(bb * 16 + h)) * 2048 + t) * 64 + d] = f2bf(v);
          else
            ovt[(((size_t)(bb * 16 + h)) * 64 + d) * 2048 + t] = f2bf(v);
        } else {
          outf[(size_t)m * N + n] = v;
        }
      }
    }
  }
}

// ---------------------------------------------------------------------------
// Flash attention (causal), transposed-score formulation, software-pipelined.
//   One wave per 16-row query tile; key blocks of 32; NO LDS.
//   S^T = K @ Q^T so each lane owns one query column t: row-max/row-sum are
//   in-lane trees + one xor-16 shuffle, and the exp'd S^T tiles pack directly
//   into the P A-fragment register layout (no LDS transpose).
//   Pipeline: V frags + next-block K frags are issued before the softmax so
//   every WMMA finds its operands resident.
//   Q,K: [B*H, T, D] bf16   Vt: [B*H, D, T] bf16   Y: [B,T,H*D] bf16
// ---------------------------------------------------------------------------
__global__ __launch_bounds__(256) void attn_flash(
    const unsigned short* __restrict__ Q,
    const unsigned short* __restrict__ Kb,
    const unsigned short* __restrict__ Vt,
    unsigned short* __restrict__ Y) {
  constexpr int T = 2048, D = 64, C = 1024;

  const int lane = threadIdx.x & 31;
  const int wave = threadIdx.x >> 5;
  const int ln15 = lane & 15;
  const int hi   = lane >> 4;
  const int kb   = hi * 8;

  const int bh   = blockIdx.y;                      // b*H + h
  const int qr0  = blockIdx.x * 128 + wave * 16;    // query tile base row
  const int tcol = qr0 + ln15;                      // this lane's query row
  const int sEnd = qr0 + 16;                        // exclusive s0 bound

  const size_t qkBase = (size_t)bh * T * D;
  const size_t vtBase = (size_t)bh * D * T;

  // Q fragments (two K-chunks covering D=64); used as the B operand of K@Q^T
  FragBF fq[2];
  {
    const unsigned short* qrow = Q + qkBase + (size_t)(qr0 + ln15) * D;
#pragma unroll
    for (int c2 = 0; c2 < 2; ++c2) {
      fq[c2].q[0] = *(const uint4*)(qrow + c2 * 32 + kb);
      fq[c2].q[1] = *(const uint4*)(qrow + c2 * 32 + kb + 16);
    }
  }

  f32x8 zero = {};
  f32x8 accY[4];
#pragma unroll
  for (int dt = 0; dt < 4; ++dt) accY[dt] = zero;

  float m_t = -3.0e38f;         // running max for this lane's query row
  float l_t = 0.0f;             // running sum
  const float scale = 0.125f;   // 1/sqrt(64)

  // K fragments, double-buffered across iterations in registers
  FragBF fk[2][2];
  {
    const unsigned short* krow0 = Kb + qkBase + (size_t)(ln15)      * D;
    const unsigned short* krow1 = Kb + qkBase + (size_t)(16 + ln15) * D;
#pragma unroll
    for (int c2 = 0; c2 < 2; ++c2) {
      fk[0][c2].q[0] = *(const uint4*)(krow0 + c2 * 32 + kb);
      fk[0][c2].q[1] = *(const uint4*)(krow0 + c2 * 32 + kb + 16);
      fk[1][c2].q[0] = *(const uint4*)(krow1 + c2 * 32 + kb);
      fk[1][c2].q[1] = *(const uint4*)(krow1 + c2 * 32 + kb + 16);
    }
  }

  for (int s0 = 0; s0 < sEnd; s0 += 32) {
    // ---- S^T tiles: A = K rows, B = Q rows (swapped operands)
    f32x8 accT[2];
#pragma unroll
    for (int st = 0; st < 2; ++st) {
      accT[st] = zero;
#pragma unroll
      for (int c2 = 0; c2 < 2; ++c2)
        accT[st] = __builtin_amdgcn_wmma_f32_16x16x32_bf16(
            false, fk[st][c2].v, false, fq[c2].v, (short)0, accT[st], false, false);
    }

    // ---- issue this block's V fragment loads NOW (hide behind softmax)
    FragBF fv[4];
#pragma unroll
    for (int dt = 0; dt < 4; ++dt) {
      const unsigned short* vrow = Vt + vtBase + (size_t)(dt * 16 + ln15) * T + s0;
      fv[dt].q[0] = *(const uint4*)(vrow + kb);
      fv[dt].q[1] = *(const uint4*)(vrow + kb + 16);
    }

    // ---- issue next block's K fragment loads (hide behind softmax + P@V)
    if (s0 + 32 < sEnd) {
      const unsigned short* krow0 = Kb + qkBase + (size_t)(s0 + 32 + ln15) * D;
      const unsigned short* krow1 = Kb + qkBase + (size_t)(s0 + 48 + ln15) * D;
#pragma unroll
      for (int c2 = 0; c2 < 2; ++c2) {
        fk[0][c2].q[0] = *(const uint4*)(krow0 + c2 * 32 + kb);
        fk[0][c2].q[1] = *(const uint4*)(krow0 + c2 * 32 + kb + 16);
        fk[1][c2].q[0] = *(const uint4*)(krow1 + c2 * 32 + kb);
        fk[1][c2].q[1] = *(const uint4*)(krow1 + c2 * 32 + kb + 16);
      }
      // prefetch next V block (d rows: lane and lane+32)
      const unsigned short* vpf = Vt + vtBase + (size_t)lane * T + s0 + 32;
      __builtin_prefetch(vpf, 0, 1);
      __builtin_prefetch(vpf + (size_t)32 * T, 0, 1);
    }

    // ---- scale + causal mask (flattened: idx = st*8 + e)
    float p[16];
#pragma unroll
    for (int idx = 0; idx < 16; ++idx) {
      int st = idx >> 3, e = idx & 7;
      int s = s0 + st * 16 + e + hi * 8;
      float x = accT[st][e] * scale;
      p[idx] = (s <= tcol) ? x : -3.0e38f;
    }

    // ---- row max: pairwise tree (depth 4) + one xor-16 shuffle
    float red[8];
#pragma unroll
    for (int i = 0; i < 8; ++i) red[i] = fmaxf(p[i], p[i + 8]);
#pragma unroll
    for (int w = 4; w; w >>= 1)
#pragma unroll
      for (int i = 0; i < w; ++i) red[i] = fmaxf(red[i], red[i + w]);
    float mx = fmaxf(red[0], __shfl_xor(red[0], 16, 32));

    float mnew  = fmaxf(m_t, mx);
    float alpha = __expf(m_t - mnew);
#pragma unroll
    for (int idx = 0; idx < 16; ++idx) p[idx] = __expf(p[idx] - mnew);

    // ---- row sum: pairwise tree + one xor-16 shuffle
#pragma unroll
    for (int i = 0; i < 8; ++i) red[i] = p[i] + p[i + 8];
#pragma unroll
    for (int w = 4; w; w >>= 1)
#pragma unroll
      for (int i = 0; i < w; ++i) red[i] += red[i + w];
    float rs = red[0] + __shfl_xor(red[0], 16, 32);

    l_t = l_t * alpha + rs;
    m_t = mnew;

    // ---- rescale accY: element e holds query row (e + 8*hi); fetch its alpha
#pragma unroll
    for (int e = 0; e < 8; ++e) {
      float ae = __shfl(alpha, e + 8 * hi, 32);
      accY[0][e] *= ae; accY[1][e] *= ae; accY[2][e] *= ae; accY[3][e] *= ae;
    }

    // ---- pack P directly into A-fragment registers (no LDS!)
    // A-frag element e -> k = (e<8 ? e : e+8) + kb: tile0 elem e, tile1 elem e-8
    FragBF fp;
#pragma unroll
    for (int e = 0; e < 8; ++e) {
      fp.u[e]     = f2bf(p[e]);
      fp.u[e + 8] = f2bf(p[e + 8]);
    }

    // ---- Y += P @ V : fv fragments already resident, WMMAs back-to-back
#pragma unroll
    for (int dt = 0; dt < 4; ++dt)
      accY[dt] = __builtin_amdgcn_wmma_f32_16x16x32_bf16(
          false, fp.v, false, fv[dt].v, (short)0, accY[dt], false, false);
  }

  // ---- normalize and store Y (bf16, [B,T,C])
  int b = bh >> 4, h = bh & 15;
#pragma unroll
  for (int e = 0; e < 8; ++e) {
    float le  = __shfl(l_t, e + 8 * hi, 32);
    float inv = 1.0f / le;
    int t = qr0 + e + hi * 8;
    unsigned short* yrow = Y + ((size_t)(b * T + t)) * C + h * D;
#pragma unroll
    for (int dt = 0; dt < 4; ++dt)
      yrow[dt * 16 + ln15] = f2bf(accY[dt][e] * inv);
  }
}

// ---------------------------------------------------------------------------
// Launch: convert -> QKV GEMM -> flash attention -> proj GEMM
// ---------------------------------------------------------------------------
extern "C" void kernel_launch(void* const* d_in, const int* in_sizes, int n_in,
                              void* d_out, int out_size, void* d_ws, size_t ws_size,
                              hipStream_t stream) {
  const float* x      = (const float*)d_in[0];   // [2,2048,1024]
  const float* w_attn = (const float*)d_in[1];   // [1024,3072]
  const float* b_attn = (const float*)d_in[2];   // [3072]
  const float* w_proj = (const float*)d_in[3];   // [1024,1024]
  const float* b_proj = (const float*)d_in[4];   // [1024]
  float* out = (float*)d_out;                    // [2,2048,1024] fp32

  const size_t MiB = 1u << 20;
  char* ws = (char*)d_ws;
  unsigned short* xb  = (unsigned short*)(ws + 0 * MiB);   //  8 MiB
  unsigned short* wab = (unsigned short*)(ws + 8 * MiB);   //  6 MiB
  unsigned short* wpb = (unsigned short*)(ws + 14 * MiB);  //  2 MiB
  unsigned short* q   = (unsigned short*)(ws + 16 * MiB);  //  8 MiB
  unsigned short* k   = (unsigned short*)(ws + 24 * MiB);  //  8 MiB
  unsigned short* vt  = (unsigned short*)(ws + 32 * MiB);  //  8 MiB
  unsigned short* y   = (unsigned short*)(ws + 40 * MiB);  //  8 MiB

  const int nx  = 2 * 2048 * 1024;     // 4194304
  const int nwa = 1024 * 3072;         // 3145728
  const int nwp = 1024 * 1024;         // 1048576

  cvt_f32_bf16<<<nx  / 1024, 256, 0, stream>>>(x,      xb,  nx);
  cvt_f32_bf16<<<nwa / 1024, 256, 0, stream>>>(w_attn, wab, nwa);
  cvt_f32_bf16<<<nwp / 1024, 256, 0, stream>>>(w_proj, wpb, nwp);

  // QKV: [4096,1024] @ [1024,3072] -> scatter q/k/vt
  gemm_bf16<3072, 1024, 0><<<dim3(24, 32), 256, 0, stream>>>(
      xb, wab, b_attn, q, k, vt, nullptr);

  // Flash attention: grid = (T/128, B*H)
  attn_flash<<<dim3(16, 32), 256, 0, stream>>>(q, k, vt, y);

  // Proj: [4096,1024] @ [1024,1024] + bias -> fp32 out
  gemm_bf16<1024, 1024, 1><<<dim3(8, 32), 256, 0, stream>>>(
      y, wpb, b_proj, nullptr, nullptr, nullptr, out);
}
